// EfficientLoFTR_73607149519184
// MI455X (gfx1250) — matrix operand order repaired and checked by reading, not compiled, not run on hardware
//
#include <hip/hip_runtime.h>
#include <hip/hip_bf16.h>
#include <math.h>

// ---------------- problem constants (from reference setup_inputs) ------------
#define NB     2
#define LDIM   4800
#define SDIM   4800
#define CDIM   256
#define THR_F  0.2f
#define BORDER 2
#define SBLK   75            // SDIM/64  (S-direction block count)
#define LBLK   75            // LDIM/64  (L-direction block count)

static constexpr float SIM_SCALE = 1.0f / (256.0f * 0.1f);  // (1/sqrt(C))^2 / TEMPERATURE

typedef __attribute__((ext_vector_type(16))) _Float16 v16h;
typedef __attribute__((ext_vector_type(8)))  _Float16 h8;
typedef __attribute__((ext_vector_type(8)))  float    v8f;
typedef __attribute__((ext_vector_type(4)))  float    f4;

// ---------------------------------------------------------------------------
// Kernel 1: sim[n,l,s] = dot(f0[n,l,:], f1[n,s,:]) * SIM_SCALE  via f16 WMMA,
// fused with per-block partial exp-sum reductions for the dual softmax.
//   block = 256 threads (8 waves), tile = 64(L) x 64(S), BK = 32
//   waves 4x2; each wave -> 16 rows x 32 cols (two 16x16 WMMA accumulators)
//   Epilogue (deterministic, no atomics):
//     rowPart[bx][n,l] = sum_{s in block cols} exp(sim)
//     colPart[by][n,s] = sum_{l in block rows} exp(sim)
// ---------------------------------------------------------------------------
__global__ __launch_bounds__(256) void sim_gemm_wmma(const float* __restrict__ f0,
                                                     const float* __restrict__ f1,
                                                     float* __restrict__ sim,
                                                     float* __restrict__ rowPart,
                                                     float* __restrict__ colPart)
{
    __shared__ __align__(16) _Float16 lA[64][40];   // padded rows: 80B stride
    __shared__ __align__(16) _Float16 lB[64][40];
    __shared__ float rowW[2][64];                   // per-wn row partials
    __shared__ float colW[4][64];                   // per-wm col partials

    const int n  = blockIdx.z;
    const int l0 = blockIdx.y * 64;
    const int s0 = blockIdx.x * 64;
    const int tid  = threadIdx.x;
    const int lane = tid & 31;
    const int wid  = tid >> 5;     // 0..7
    const int wm   = wid >> 1;     // 0..3  (16-row band)
    const int wn   = wid & 1;      // 0..1  (32-col band)

    // global->LDS staging: each thread loads 8 consecutive floats of one row
    const int ldrow = tid >> 2;          // 0..63
    const int ldcol = (tid & 3) * 8;     // 0,8,16,24

    const float* aG = f0 + (size_t)n * LDIM * CDIM + (size_t)(l0 + ldrow) * CDIM + ldcol;
    const float* bG = f1 + (size_t)n * SDIM * CDIM + (size_t)(s0 + ldrow) * CDIM + ldcol;

    // fragment addressing (CDNA5 16-bit WMMA layouts, wave32)
    const int grp = lane >> 4;           // lane half
    const int hid = lane & 15;
    const int am  = wm * 16 + hid;       // A row (M) held by this lane
    const int bn  = wn * 32 + hid;       // B col (N) held by this lane (tile 0)

    v8f acc0 = {};
    v8f acc1 = {};

    // register double-buffer: preload K-block 0
    f4 a0 = *(const f4*)(aG);
    f4 a1 = *(const f4*)(aG + 4);
    f4 b0 = *(const f4*)(bG);
    f4 b1 = *(const f4*)(bG + 4);

    for (int kk = 0; kk < CDIM; kk += 32) {
        _Float16* pa = &lA[ldrow][ldcol];
        _Float16* pb = &lB[ldrow][ldcol];
#pragma unroll
        for (int i = 0; i < 4; ++i) {
            pa[i]     = (_Float16)a0[i];
            pa[i + 4] = (_Float16)a1[i];
            pb[i]     = (_Float16)b0[i];
            pb[i + 4] = (_Float16)b1[i];
        }
        __syncthreads();

        // issue next K-block global loads; they overlap the WMMA work below
        if (kk + 32 < CDIM) {
            a0 = *(const f4*)(aG + kk + 32);
            a1 = *(const f4*)(aG + kk + 36);
            b0 = *(const f4*)(bG + kk + 32);
            b1 = *(const f4*)(bG + kk + 36);
            if (kk + 64 < CDIM) {
                __builtin_prefetch(aG + kk + 64, 0, 1);   // global_prefetch_b8
                __builtin_prefetch(bG + kk + 64, 0, 1);
            }
        }

        // A 16x32: lane<16 holds K 0..15 of row M=hid; lane>=16 holds K 16..31
        h8 alo  = *(const h8*)&lA[am][grp * 16];
        h8 ahi  = *(const h8*)&lA[am][grp * 16 + 8];
        // B 32x16: lane<16 holds K 0..15 of col N=hid; lane>=16 holds K 16..31
        h8 b0lo = *(const h8*)&lB[bn][grp * 16];
        h8 b0hi = *(const h8*)&lB[bn][grp * 16 + 8];
        h8 b1lo = *(const h8*)&lB[bn + 16][grp * 16];
        h8 b1hi = *(const h8*)&lB[bn + 16][grp * 16 + 8];

        v16h aF, bF0, bF1;
#pragma unroll
        for (int i = 0; i < 8; ++i) {
            aF[i]  = alo[i];  aF[i + 8]  = ahi[i];
            bF0[i] = b0lo[i]; bF0[i + 8] = b0hi[i];
            bF1[i] = b1lo[i]; bF1[i + 8] = b1hi[i];
        }

        acc0 = __builtin_amdgcn_wmma_f32_16x16x32_f16(false, aF, false, bF0,
                                                      (short)0, acc0, false, false);
        acc1 = __builtin_amdgcn_wmma_f32_16x16x32_f16(false, aF, false, bF1,
                                                      (short)0, acc1, false, false);
        __syncthreads();
    }

    // ---- epilogue: store sim + deterministic partial exp-sum reductions ----
    // D layout: lanes 0-15 -> M = r (+row base), N = hid; lanes 16-31 -> M = 8 + r
    float* oB = sim + (size_t)n * LDIM * SDIM;
    const int orow  = l0 + wm * 16 + grp * 8;
    const int ocol0 = s0 + wn * 32 + hid;

    float e0[8], e1[8];
    float cs0 = 0.0f, cs1 = 0.0f;
#pragma unroll
    for (int r = 0; r < 8; ++r) {
        float v0 = acc0[r] * SIM_SCALE;
        float v1 = acc1[r] * SIM_SCALE;
        oB[(size_t)(orow + r) * SDIM + ocol0]      = v0;
        oB[(size_t)(orow + r) * SDIM + ocol0 + 16] = v1;
        e0[r] = __expf(v0);
        e1[r] = __expf(v1);
        cs0 += e0[r];
        cs1 += e1[r];
    }

    // column sums: combine the two lane-halves (rows 0-7 with rows 8-15)
    cs0 += __shfl_xor(cs0, 16, 32);
    cs1 += __shfl_xor(cs1, 16, 32);
    if (lane < 16) {
        colW[wm][wn * 32 + hid]      = cs0;
        colW[wm][wn * 32 + 16 + hid] = cs1;
    }

    // row sums: reduce across the 16 lanes of each half (different columns)
    float rv[8];
#pragma unroll
    for (int r = 0; r < 8; ++r) rv[r] = e0[r] + e1[r];
#pragma unroll
    for (int off = 1; off < 16; off <<= 1) {
#pragma unroll
        for (int r = 0; r < 8; ++r) rv[r] += __shfl_xor(rv[r], off, 32);
    }
    if (hid == 0) {
#pragma unroll
        for (int r = 0; r < 8; ++r) rowW[wn][wm * 16 + grp * 8 + r] = rv[r];
    }
    __syncthreads();

    if (tid < 64) {
        // unique (block, index) writers -> deterministic
        rowPart[(size_t)blockIdx.x * (NB * LDIM) + (size_t)n * LDIM + (l0 + tid)] =
            rowW[0][tid] + rowW[1][tid];
        colPart[(size_t)blockIdx.y * (NB * SDIM) + (size_t)n * SDIM + (s0 + tid)] =
            (colW[0][tid] + colW[1][tid]) + (colW[2][tid] + colW[3][tid]);
    }
}

// ---------------------------------------------------------------------------
// Kernel 2: deterministic fixed-order reduction of the partial exp-sums
//   idx in [0, NB*LDIM)              -> rsum
//   idx in [NB*LDIM, NB*LDIM+NB*SDIM)-> csum
// ---------------------------------------------------------------------------
__global__ __launch_bounds__(256) void reduce_parts(const float* __restrict__ rowPart,
                                                    const float* __restrict__ colPart,
                                                    float* __restrict__ rsum,
                                                    float* __restrict__ csum)
{
    const int NL = NB * LDIM;
    const int NS = NB * SDIM;
    int idx = blockIdx.x * 256 + threadIdx.x;
    if (idx < NL) {
        float s = 0.0f;
        for (int b = 0; b < SBLK; ++b) s += rowPart[(size_t)b * NL + idx];
        rsum[idx] = s;
    } else if (idx < NL + NS) {
        int j = idx - NL;
        float s = 0.0f;
        for (int b = 0; b < LBLK; ++b) s += colPart[(size_t)b * NS + j];
        csum[j] = s;
    }
}

// ---------------------------------------------------------------------------
// Kernel 3: conf = (exp(x)/rsum) * (exp(x)/csum), in place over sim;
//           per-row max of conf (plain store, one block per row)
// ---------------------------------------------------------------------------
__global__ __launch_bounds__(256) void conf_rows(float* __restrict__ conf,
                                                 const float* __restrict__ rsum,
                                                 const float* __restrict__ csum,
                                                 float* __restrict__ rowmaxc)
{
    const int l = blockIdx.x, n = blockIdx.y, t = threadIdx.x;
    float* row = conf + ((size_t)n * LDIM + l) * SDIM;
    const float inv_rs = 1.0f / rsum[n * LDIM + l];
    const float* cs = csum + n * SDIM;
    float lmax = 0.0f;
    for (int j = t; j < SDIM; j += 256) {
        float ex = __expf(row[j]);
        float c  = (ex * inv_rs) * (ex / cs[j]);
        row[j] = c;
        lmax = fmaxf(lmax, c);
    }
    __shared__ float sm[256];
    sm[t] = lmax;
    __syncthreads();
    for (int off = 128; off > 0; off >>= 1) {
        if (t < off) sm[t] = fmaxf(sm[t], sm[t + off]);
        __syncthreads();
    }
    if (t == 0) rowmaxc[n * LDIM + l] = sm[0];
}

// ---------------------------------------------------------------------------
// Kernel 4: per-column max of conf (64 columns per block, 4 row-groups)
// ---------------------------------------------------------------------------
__global__ __launch_bounds__(256) void colmax_conf(const float* __restrict__ conf,
                                                   float* __restrict__ colmaxc)
{
    const int n = blockIdx.y, t = threadIdx.x;
    const int c  = t & 63;
    const int rg = t >> 6;
    const int s_idx = blockIdx.x * 64 + c;
    const float* base = conf + (size_t)n * LDIM * SDIM;
    float m = 0.0f;
    for (int r = rg; r < LDIM; r += 4)
        m = fmaxf(m, base[(size_t)r * SDIM + s_idx]);
    __shared__ float sm[256];
    sm[t] = m;
    __syncthreads();
    if (t < 64)
        colmaxc[n * SDIM + s_idx] =
            fmaxf(fmaxf(sm[t], sm[t + 64]), fmaxf(sm[t + 128], sm[t + 192]));
}

// ---------------------------------------------------------------------------
// Kernel 5: threshold + border + mutual-NN, per-row match extraction
// ---------------------------------------------------------------------------
__global__ __launch_bounds__(256) void match_extract(const float* __restrict__ conf,
                                                     const float* __restrict__ rowmaxc,
                                                     const float* __restrict__ colmaxc,
                                                     const int* __restrict__ p_h0c,
                                                     const int* __restrict__ p_w0c,
                                                     const int* __restrict__ p_h1c,
                                                     const int* __restrict__ p_w1c,
                                                     const int* __restrict__ p_hi,
                                                     float* __restrict__ oValid,
                                                     float* __restrict__ oJ,
                                                     float* __restrict__ oMconf,
                                                     float* __restrict__ oK0,
                                                     float* __restrict__ oK1)
{
    const int l = blockIdx.x, n = blockIdx.y, t = threadIdx.x;
    const int h0c = *p_h0c, w0c = *p_w0c, h1c = *p_h1c, w1c = *p_w1c, h_i = *p_hi;
    const int r0 = l / w0c, c0 = l % w0c;
    const bool vl = (r0 >= BORDER) && (r0 < h0c - BORDER) &&
                    (c0 >= BORDER) && (c0 < w0c - BORDER);
    const float rmx = rowmaxc[n * LDIM + l];
    const float* crow = conf + ((size_t)n * LDIM + l) * SDIM;
    const float* cmx  = colmaxc + n * SDIM;

    int best = 0x7FFFFFFF;
    for (int j = t; j < SDIM; j += 256) {
        float c = crow[j];
        int r1 = j / w1c, c1 = j % w1c;
        bool vs = (r1 >= BORDER) && (r1 < h1c - BORDER) &&
                  (c1 >= BORDER) && (c1 < w1c - BORDER);
        bool msk = (c > THR_F) && vl && vs && (c == rmx) && (c == cmx[j]);
        if (msk && j < best) best = j;
    }
    __shared__ int sj[256];
    sj[t] = best;
    __syncthreads();
    for (int off = 128; off > 0; off >>= 1) {
        if (t < off) sj[t] = min(sj[t], sj[t + off]);
        __syncthreads();
    }
    if (t == 0) {
        const bool any = (sj[0] != 0x7FFFFFFF);
        const int  j   = any ? sj[0] : 0;            // argmax-of-bool-mask semantics
        const float mc = any ? crow[j] : 0.0f;
        const int idx = n * LDIM + l;
        oValid[idx] = (any && (mc != 0.0f)) ? 1.0f : 0.0f;
        oJ[idx]     = (float)j;
        oMconf[idx] = mc;
        const float scale = (float)h_i / (float)h0c;  // == 8.0
        oK0[idx * 2 + 0] = (float)(l % w0c) * scale;
        oK0[idx * 2 + 1] = (float)(l / w0c) * scale;
        oK1[idx * 2 + 0] = (float)(j % w1c) * scale;
        oK1[idx * 2 + 1] = (float)(j / w1c) * scale;
    }
}

// ---------------------------------------------------------------------------
extern "C" void kernel_launch(void* const* d_in, const int* in_sizes, int n_in,
                              void* d_out, int out_size, void* d_ws, size_t ws_size,
                              hipStream_t stream)
{
    (void)in_sizes; (void)n_in; (void)out_size; (void)ws_size;

    const float* f0 = (const float*)d_in[0];   // [2,4800,256]
    const float* f1 = (const float*)d_in[1];   // [2,4800,256]
    const int* h0c = (const int*)d_in[2];
    const int* w0c = (const int*)d_in[3];
    const int* h1c = (const int*)d_in[4];
    const int* w1c = (const int*)d_in[5];
    const int* h_i = (const int*)d_in[6];

    float* out = (float*)d_out;
    const size_t NL = (size_t)NB * LDIM;           // 9600
    const size_t NS = (size_t)NB * SDIM;           // 9600
    const size_t confElems = (size_t)NB * LDIM * SDIM;

    // output layout (flat concat, reference return order)
    float* conf   = out;
    float* oValid = out + confElems;
    float* oJ     = oValid + NL;
    float* oMconf = oJ + NL;
    float* oK0    = oMconf + NL;
    float* oK1    = oK0 + 2 * NL;

    // workspace: softmax sums, maxima, per-block partial exp-sums (~5.9 MB)
    float* ws      = (float*)d_ws;
    float* rsum    = ws;                       // NL
    float* csum    = rsum + NL;                // NS
    float* rowmaxc = csum + NS;                // NL
    float* colmaxc = rowmaxc + NL;             // NS
    float* rowPart = colmaxc + NS;             // SBLK * NL
    float* colPart = rowPart + (size_t)SBLK * NL;  // LBLK * NS

    // 1) WMMA GEMM -> sim in conf region, fused partial exp-sums
    sim_gemm_wmma<<<dim3(SDIM / 64, LDIM / 64, NB), 256, 0, stream>>>(
        f0, f1, conf, rowPart, colPart);
    // 2) deterministic reduction of partials -> rsum / csum
    reduce_parts<<<dim3((unsigned)((NL + NS + 255) / 256)), 256, 0, stream>>>(
        rowPart, colPart, rsum, csum);
    // 3) conf = p_row * p_col (in place) + row maxima
    conf_rows<<<dim3(LDIM, NB), 256, 0, stream>>>(conf, rsum, csum, rowmaxc);
    // 4) column maxima
    colmax_conf<<<dim3(SDIM / 64, NB), 256, 0, stream>>>(conf, colmaxc);
    // 5) match extraction
    match_extract<<<dim3(LDIM, NB), 256, 0, stream>>>(conf, rowmaxc, colmaxc,
                                                      h0c, w0c, h1c, w1c, h_i,
                                                      oValid, oJ, oMconf, oK0, oK1);
}
